// ConvAttention_10496900071844
// MI455X (gfx1250) — matrix operand order repaired and checked by reading, not compiled, not run on hardware
//
#include <hip/hip_runtime.h>

typedef unsigned short u16;
typedef unsigned int   u32;
typedef __attribute__((ext_vector_type(16))) __bf16 v16bf;
typedef __attribute__((ext_vector_type(8)))  float  v8f;

#define NB     8
#define NC     192
#define NH     28
#define NTOK   784
#define NHEADS 6
#define CHD    32
#define ATT_SCALE 0.07216878364870323f   // 192^-0.5

__device__ __forceinline__ u16 f2bf(float f) {
  union { float f; u32 u; } c; c.f = f;
  u32 r = c.u + 0x7FFFu + ((c.u >> 16) & 1u);   // round-to-nearest-even
  return (u16)(r >> 16);
}

union FragBF {
  v16bf bf;
  uint4 q[2];
  u16   s[16];
};

__device__ __forceinline__ v8f wmma_bf16(v16bf a, v16bf b, v8f c) {
  return __builtin_amdgcn_wmma_f32_16x16x32_bf16(false, a, false, b, (short)0, c,
                                                 false, false);
}

// ---------------------------------------------------------------- weights cvt
__global__ void cvt_w_kernel(const float* __restrict__ pq, const float* __restrict__ pk,
                             const float* __restrict__ pv, const float* __restrict__ po,
                             u16* __restrict__ wq, u16* __restrict__ wk,
                             u16* __restrict__ wv, u16* __restrict__ wo) {
  int i = blockIdx.x * blockDim.x + threadIdx.x;
  if (i < NC * NC) {
    wq[i] = f2bf(pq[i]); wk[i] = f2bf(pk[i]);
    wv[i] = f2bf(pv[i]); wo[i] = f2bf(po[i]);
  }
}

// ------------------------------------------------------- depthwise conv + BN
// Tiled: one block = (batch, 32-channel group, 4-row strip). The 6x28x32 f32
// input halo tile is staged into LDS with CDNA5 async global->LDS copies
// (each input element is reused up to 9x by the 3x3 window).
// x: (B, N, C) token-major f32.  act out: (B, N, C) token-major bf16.
__global__ void __launch_bounds__(256) dwconv_bn_kernel(
    const float* __restrict__ x, const float* __restrict__ w,
    const float* __restrict__ g, const float* __restrict__ bb,
    const float* __restrict__ mm, const float* __restrict__ vv,
    u16* __restrict__ act) {
  __shared__ __align__(16) float xs[6][NH][32];   // rows i0-1 .. i0+4
  int tid = threadIdx.x;
  int b  = blockIdx.x / 42;
  int cg = (blockIdx.x / 7) % 6;
  int st = blockIdx.x % 7;
  int i0 = st * 4;
  int c0 = cg * 32;

  // ---- async stage: 6 rows x 28 cols x (8 chunks of 4 floats) = 1344 b128s
  for (int s = tid; s < 6 * NH * 8; s += 256) {
    int chunk = s & 7;
    int col   = (s >> 3) % NH;
    int rl    = s / (8 * NH);
    int gi    = i0 - 1 + rl;
    float* lp = &xs[rl][col][chunk * 4];
    if (gi >= 0 && gi < NH) {
      const float* gp = x + ((size_t)(b * NTOK + gi * NH + col)) * NC + c0 + chunk * 4;
      unsigned lds_off = (unsigned)(uintptr_t)lp;
      asm volatile("global_load_async_to_lds_b128 %0, %1, off"
                   :: "v"(lds_off), "v"((const void*)gp) : "memory");
    } else {
      *(float4*)lp = make_float4(0.f, 0.f, 0.f, 0.f);   // zero-pad halo
    }
  }
  asm volatile("s_wait_asynccnt 0x0" ::: "memory");
  __syncthreads();

  // ---- compute: 4 rows x 28 cols x 32 ch = 3584 outputs, 14 per thread
  for (int s = tid; s < 4 * NH * 32; s += 256) {
    int c   = s & 31;
    int col = (s >> 5) % NH;
    int il  = s / (32 * NH);
    int gi  = i0 + il;
    int gc  = c0 + c;
    float acc = 0.f;
    #pragma unroll
    for (int di = 0; di < 3; ++di) {
      #pragma unroll
      for (int dj = -1; dj <= 1; ++dj) {
        int cj = col + dj;
        if (cj >= 0 && cj < NH)
          acc += w[gc * 9 + di * 3 + (dj + 1)] * xs[il + di][cj][c];
      }
    }
    float inv = g[gc] * rsqrtf(vv[gc] + 1e-5f);
    float out = acc * inv + (bb[gc] - mm[gc] * inv);
    act[((size_t)(b * NTOK + gi * NH + col)) * NC + gc] = f2bf(out);
  }
}

// ------------------------------------------------------------ projection GEMM
// out[o,n] = sum_c W[o,c] * act[n,c] + bias[o] per batch.
// act token-major: the B fragment (16 consecutive K=c per lane at fixed n) is
// a direct 32B global load; A (weights row-major) uses the same contiguous
// split ({0..15} / {16..31}). Zero LDS, zero barriers.
// One wave = 32x16 output tile (2 A-frags share one B-frag). K=192 unrolled.
// kmode 0: dst[((b*6+head)*784+n)*32+ch]  (token-major: q, k)
// kmode 1: dst[((b*6+head)*32+ch)*784+n]  (channel-major: v)
__global__ void __launch_bounds__(256) proj_gemm_kernel(const u16* __restrict__ W,
    const u16* __restrict__ act, const float* __restrict__ bias,
    u16* __restrict__ dst, int kmode) {
  int lane = threadIdx.x & 31;
  int wid  = blockIdx.x * 8 + (threadIdx.x >> 5);
  int b  = wid / (6 * 49);
  int mt = (wid / 49) % 6;
  int nt = wid % 49;
  int o0 = mt * 32, n0 = nt * 16;
  int half = lane >> 4;
  int l16  = lane & 15;
  const u16* arow0 = W + (o0 + l16) * NC + half * 16;       // A rows o0..o0+15
  const u16* arow1 = arow0 + 16 * NC;                       // A rows o0+16..o0+31
  const u16* brow  = act + ((size_t)(b * NTOK) + n0 + l16) * NC + half * 16;
  v8f acc0 = {}, acc1 = {};
  #pragma unroll
  for (int k0 = 0; k0 < NC; k0 += 32) {
    FragBF fa0, fa1, fb;
    fa0.q[0] = *(const uint4*)(arow0 + k0);
    fa0.q[1] = *(const uint4*)(arow0 + k0 + 8);
    fa1.q[0] = *(const uint4*)(arow1 + k0);
    fa1.q[1] = *(const uint4*)(arow1 + k0 + 8);
    fb.q[0]  = *(const uint4*)(brow + k0);
    fb.q[1]  = *(const uint4*)(brow + k0 + 8);
    acc0 = wmma_bf16(fa0.bf, fb.bf, acc0);
    acc1 = wmma_bf16(fa1.bf, fb.bf, acc1);
  }
  int n = n0 + l16;
  #pragma unroll
  for (int half2 = 0; half2 < 2; ++half2) {
    v8f acc = half2 ? acc1 : acc0;
    #pragma unroll
    for (int r = 0; r < 8; ++r) {
      int o = o0 + half2 * 16 + r + half * 8;
      float val = acc[r] + bias[o];
      int head = o >> 5, chn = o & 31;
      if (kmode == 0)
        dst[((b * NHEADS + head) * NTOK + n) * CHD + chn] = f2bf(val);
      else
        dst[((b * NHEADS + head) * CHD + chn) * NTOK + n] = f2bf(val);
    }
  }
}

// ------------------------------------------------------------ flash attention
// One wave per (b, head, 16-query tile), 8 waves/block. Online softmax over
// 784 keys in 25 blocks of 32.
// Orientation: A = Q (rows = query tokens), B = K^T (cols = key tokens), so
// the S/O D-fragments and the running stats all index rows as (r + 8*half).
// Layouts give direct 128-bit fragment loads: q,k token-major (n,ch);
// v channel-major (ch,n). Only the P (D->A layout) transpose uses LDS.
// attout: (B, 784, 192) token-major bf16, channel permuted c' = ch*6 + head.
__global__ void __launch_bounds__(256) attention_kernel(const u16* __restrict__ qb,
    const u16* __restrict__ kb, const u16* __restrict__ vb, u16* __restrict__ attout) {
  __shared__ __align__(16) u16 Ps[8][16][32];   // per-wave P tile [query][key]
  int lane = threadIdx.x & 31;
  int w8   = threadIdx.x >> 5;
  int wid  = blockIdx.x * 8 + w8;
  int l16  = lane & 15;
  int half = lane >> 4;
  int bh = wid / 49;     // b*6 + h
  int qt = wid % 49;
  int n0 = qt * 16;

  // Q A-fragment (16 query rows x 32 ch), held for the whole kernel.
  // A-layout: lane<16 holds K={0..15}, lane>=16 holds K={16..31}.
  FragBF fq;
  const u16* qp = qb + (bh * NTOK + n0 + l16) * CHD + half * 16;
  fq.q[0] = *(const uint4*)qp;
  fq.q[1] = *(const uint4*)(qp + 8);

  v8f Oa = {}, Ob = {};
  float mrow[8], lrow[8];
  #pragma unroll
  for (int r = 0; r < 8; ++r) { mrow[r] = -3.0e38f; lrow[r] = 0.f; }

  const u16* kbase = kb + bh * NTOK * CHD;                 // token-major
  const u16* vrow0 = vb + (bh * CHD + l16) * NTOK;         // channel l16
  const u16* vrow1 = vb + (bh * CHD + 16 + l16) * NTOK;    // channel 16+l16

  for (int mb = 0; mb < 25; ++mb) {
    int m0 = mb * 32;
    bool full = (mb < 24);   // last block: only 16 valid keys (784 = 24*32+16)
    // ---- B-frags for S: lane = key column m = l16 (per sub-tile), K rows =
    //      16 consecutive channels (half*16..) -> one contiguous 32B load.
    FragBF bk0, bk1;
    {
      const u16* p0 = kbase + (m0 + l16) * CHD + half * 16;
      // clamp second sub-tile to valid memory when not full (result unused)
      const u16* p1 = kbase + ((full ? m0 + 16 : m0) + l16) * CHD + half * 16;
      bk0.q[0] = *(const uint4*)p0; bk0.q[1] = *(const uint4*)(p0 + 8);
      bk1.q[0] = *(const uint4*)p1; bk1.q[1] = *(const uint4*)(p1 + 8);
      if (full) __builtin_prefetch(kbase + (m0 + 32 + l16) * CHD, 0, 1);
    }
    v8f S0 = {}, S1 = {};
    S0 = wmma_bf16(fq.bf, bk0.bf, S0);   // rows = query, cols = key
    S1 = wmma_bf16(fq.bf, bk1.bf, S1);
    // ---- online softmax per query row (r + 8*half); the 16 key columns of a
    //      row live in the 16 lanes of this half-wave -> xor 1/2/4/8 reduce.
    #pragma unroll
    for (int r = 0; r < 8; ++r) {
      float s0 = S0[r] * ATT_SCALE;
      float s1 = full ? S1[r] * ATT_SCALE : -3.0e38f;
      float bm = fmaxf(s0, s1);
      bm = fmaxf(bm, __shfl_xor(bm, 1));
      bm = fmaxf(bm, __shfl_xor(bm, 2));
      bm = fmaxf(bm, __shfl_xor(bm, 4));
      bm = fmaxf(bm, __shfl_xor(bm, 8));
      float nm = fmaxf(mrow[r], bm);
      float al = __expf(mrow[r] - nm);
      float p0e = __expf(s0 - nm);
      float p1e = full ? __expf(s1 - nm) : 0.f;
      float rs = p0e + p1e;
      rs += __shfl_xor(rs, 1);
      rs += __shfl_xor(rs, 2);
      rs += __shfl_xor(rs, 4);
      rs += __shfl_xor(rs, 8);
      lrow[r] = lrow[r] * al + rs;
      mrow[r] = nm;
      Oa[r] *= al; Ob[r] *= al;
      int row = r + half * 8;
      Ps[w8][row][l16]      = f2bf(p0e);
      Ps[w8][row][16 + l16] = f2bf(p1e);
    }
    __syncthreads();
    // ---- O += P V^T : P A-frag from LDS; V B-frags direct channel-major
    FragBF fp, bv0, bv1;
    { const u16* p = &Ps[w8][l16][half * 16];
      fp.q[0] = *(const uint4*)p; fp.q[1] = *(const uint4*)(p + 8); }
    {
      int moff = m0 + (full ? half * 16 : 0);   // clamp: padded P cols are 0
      const u16* p0 = vrow0 + moff;
      const u16* p1 = vrow1 + moff;
      bv0.q[0] = *(const uint4*)p0; bv0.q[1] = *(const uint4*)(p0 + 8);
      bv1.q[0] = *(const uint4*)p1; bv1.q[1] = *(const uint4*)(p1 + 8);
      if (full) { __builtin_prefetch(vrow0 + m0 + 32, 0, 1);
                  __builtin_prefetch(vrow1 + m0 + 32, 0, 1); }
    }
    Oa = wmma_bf16(fp.bf, bv0.bf, Oa);
    Ob = wmma_bf16(fp.bf, bv1.bf, Ob);
    __syncthreads();
  }
  // ---- epilogue: normalize with this row's own stats, permute channels
  int b = bh / NHEADS, h = bh % NHEADS;
  #pragma unroll
  for (int r = 0; r < 8; ++r) {
    float inv = 1.0f / lrow[r];
    int n = n0 + r + half * 8;
    attout[((size_t)(b * NTOK) + n) * NC + l16 * NHEADS + h]        = f2bf(Oa[r] * inv);
    attout[((size_t)(b * NTOK) + n) * NC + (16 + l16) * NHEADS + h] = f2bf(Ob[r] * inv);
  }
}

// ------------------------------------------------------------- output GEMM
// y[b,o,g,n] = sum_c' po_w[o,c'] * attout[b,n,c'] + po_b[o]; 4 identical
// g-copies. attout token-major -> direct-load B fragments, zero LDS.
__global__ void __launch_bounds__(256) out_gemm_kernel(const u16* __restrict__ W,
    const u16* __restrict__ act, const float* __restrict__ bias, float* __restrict__ y) {
  int lane = threadIdx.x & 31;
  int wid  = blockIdx.x * 8 + (threadIdx.x >> 5);
  int b  = wid / (6 * 49);
  int mt = (wid / 49) % 6;
  int nt = wid % 49;
  int o0 = mt * 32, n0 = nt * 16;
  int half = lane >> 4;
  int l16  = lane & 15;
  const u16* arow0 = W + (o0 + l16) * NC + half * 16;
  const u16* arow1 = arow0 + 16 * NC;
  const u16* brow  = act + ((size_t)(b * NTOK) + n0 + l16) * NC + half * 16;
  v8f acc0 = {}, acc1 = {};
  #pragma unroll
  for (int k0 = 0; k0 < NC; k0 += 32) {
    FragBF fa0, fa1, fb;
    fa0.q[0] = *(const uint4*)(arow0 + k0);
    fa0.q[1] = *(const uint4*)(arow0 + k0 + 8);
    fa1.q[0] = *(const uint4*)(arow1 + k0);
    fa1.q[1] = *(const uint4*)(arow1 + k0 + 8);
    fb.q[0]  = *(const uint4*)(brow + k0);
    fb.q[1]  = *(const uint4*)(brow + k0 + 8);
    acc0 = wmma_bf16(fa0.bf, fb.bf, acc0);
    acc1 = wmma_bf16(fa1.bf, fb.bf, acc1);
  }
  int n = n0 + l16;
  #pragma unroll
  for (int half2 = 0; half2 < 2; ++half2) {
    v8f acc = half2 ? acc1 : acc0;
    #pragma unroll
    for (int r = 0; r < 8; ++r) {
      int o = o0 + half2 * 16 + r + half * 8;
      float val = acc[r] + bias[o];
      #pragma unroll
      for (int g2 = 0; g2 < 4; ++g2)
        y[(((size_t)(b * NC) + o) * 4 + g2) * NTOK + n] = val;
    }
  }
}

// ----------------------------------------------------------------- launcher
extern "C" void kernel_launch(void* const* d_in, const int* in_sizes, int n_in,
                              void* d_out, int out_size, void* d_ws, size_t ws_size,
                              hipStream_t stream) {
  (void)in_sizes; (void)n_in; (void)out_size; (void)ws_size;
  const float* x = (const float*)d_in[0];
  const float* dww[3] = { (const float*)d_in[3],  (const float*)d_in[10], (const float*)d_in[17] };
  const float* bng[3] = { (const float*)d_in[4],  (const float*)d_in[11], (const float*)d_in[18] };
  const float* bnb[3] = { (const float*)d_in[5],  (const float*)d_in[12], (const float*)d_in[19] };
  const float* bnm[3] = { (const float*)d_in[6],  (const float*)d_in[13], (const float*)d_in[20] };
  const float* bnv[3] = { (const float*)d_in[7],  (const float*)d_in[14], (const float*)d_in[21] };
  const float* pw[3]  = { (const float*)d_in[8],  (const float*)d_in[15], (const float*)d_in[22] };
  const float* pb[3]  = { (const float*)d_in[9],  (const float*)d_in[16], (const float*)d_in[23] };
  const float* po_w = (const float*)d_in[24];
  const float* po_b = (const float*)d_in[25];

  u16* ws = (u16*)d_ws;
  const size_t WSZ = (size_t)NC * NC;           // 36864
  const size_t ASZ = (size_t)NB * NC * NTOK;    // 1204224
  u16* wq   = ws;
  u16* wk   = wq + WSZ;
  u16* wv   = wk + WSZ;
  u16* wo   = wv + WSZ;
  u16* actq = wo + WSZ;
  u16* actk = actq + ASZ;
  u16* actv = actk + ASZ;
  u16* qbuf = actv + ASZ;
  u16* kbuf = qbuf + ASZ;
  u16* vbuf = kbuf + ASZ;
  u16* atto = vbuf + ASZ;   // attout (B, 784, 192) token-major

  cvt_w_kernel<<<(NC * NC + 255) / 256, 256, 0, stream>>>(pw[0], pw[1], pw[2], po_w,
                                                          wq, wk, wv, wo);
  u16* acts[3] = { actq, actk, actv };
  for (int p = 0; p < 3; ++p)
    dwconv_bn_kernel<<<NB * 6 * 7, 256, 0, stream>>>(
        x, dww[p], bng[p], bnb[p], bnm[p], bnv[p], acts[p]);

  // 8 batches * 6 M-tiles(32) * 49 N-tiles(16) = 2352 waves, 8 waves/block
  proj_gemm_kernel<<<294, 256, 0, stream>>>(wq, actq, pb[0], qbuf, 0);
  proj_gemm_kernel<<<294, 256, 0, stream>>>(wk, actk, pb[1], kbuf, 0);
  proj_gemm_kernel<<<294, 256, 0, stream>>>(wv, actv, pb[2], vbuf, 1);

  // 8 * 6 heads * 49 query tiles = 2352 waves, 8 waves/block
  attention_kernel<<<294, 256, 0, stream>>>(qbuf, kbuf, vbuf, atto);

  out_gemm_kernel<<<294, 256, 0, stream>>>(wo, atto, po_b, (float*)d_out);
}
